// CrossAttention_84018150244741
// MI455X (gfx1250) — matrix operand order repaired and checked by reading, not compiled, not run on hardware
//
#include <hip/hip_runtime.h>
#include <hip/hip_bf16.h>

typedef _Float16 f16;
typedef __attribute__((ext_vector_type(16))) _Float16 v16h;
typedef __attribute__((ext_vector_type(8)))  _Float16 v8h;
typedef __attribute__((ext_vector_type(8)))  float    v8f;

constexpr int BB  = 2;
constexpr int LQ  = 2048;
constexpr int LKV = 4096;
constexpr int DIM = 2048;
constexpr int NH  = 16;
constexpr int HD  = 128;
constexpr float QK_SCALE = 0.08838834764831845f;  // 1/sqrt(128)
constexpr float LN_EPS   = 1e-6f;

// ---------------------------------------------------------------------------
// WMMA fragment loaders (wave32; contiguous k order, same for A and B)
// A: 16x32 f16.  lane<16: row=lane, K 0..15; lane>=16:
//    row=lane-16, K 16..31.  16 contiguous K per lane (matches B).
// B: 32x16 f16 from row-major W[n,k] (B[k][n] = W[n][k]).  lane&15 = n,
//    lane>>4 selects K half: 16 contiguous halfs (32B) per lane.
// Overloads convert f32 sources to f16 on the fly (saves staging buffers
// and HBM traffic; accumulate stays f32).
// ---------------------------------------------------------------------------
__device__ __forceinline__ v16h load_a(const f16* __restrict__ A, int lda,
                                       int row0, int k0, int lane) {
  int r    = row0 + (lane & 15);
  int koff = k0 + ((lane >> 4) << 4);
  const f16* p = A + (size_t)r * lda + koff;
  v8h a0 = *(const v8h*)p;
  v8h a1 = *(const v8h*)(p + 8);
  v16h a;
#pragma unroll
  for (int i = 0; i < 8; ++i) { a[i] = a0[i]; a[8 + i] = a1[i]; }
  return a;
}

__device__ __forceinline__ v16h load_a(const float* __restrict__ A, int lda,
                                       int row0, int k0, int lane) {
  int r    = row0 + (lane & 15);
  int koff = k0 + ((lane >> 4) << 4);
  const float* p = A + (size_t)r * lda + koff;
  v8f a0 = *(const v8f*)p;
  v8f a1 = *(const v8f*)(p + 8);
  v16h a;
#pragma unroll
  for (int i = 0; i < 8; ++i) { a[i] = (f16)a0[i]; a[8 + i] = (f16)a1[i]; }
  return a;
}

__device__ __forceinline__ v16h load_b(const f16* __restrict__ W, int ldb,
                                       int n0, int k0, int lane) {
  int n  = n0 + (lane & 15);
  int kk = k0 + ((lane >> 4) << 4);
  return *(const v16h*)(W + (size_t)n * ldb + kk);
}

__device__ __forceinline__ v16h load_b(const float* __restrict__ W, int ldb,
                                       int n0, int k0, int lane) {
  int n  = n0 + (lane & 15);
  int kk = k0 + ((lane >> 4) << 4);
  const float* p = W + (size_t)n * ldb + kk;
  v8f b0 = *(const v8f*)p;
  v8f b1 = *(const v8f*)(p + 8);
  v16h b;
#pragma unroll
  for (int i = 0; i < 8; ++i) { b[i] = (f16)b0[i]; b[8 + i] = (f16)b1[i]; }
  return b;
}

#define WMMA_F16(a, b, c) \
  __builtin_amdgcn_wmma_f32_16x16x32_f16(false, (a), false, (b), (short)0, (c), false, false)

// ---------------------------------------------------------------------------
// Core 32x64 output tile per wave: 2 A fragments x 4 B tiles -> 8 accums.
// All 6 fragment loads of a k-step are issued before the 8 WMMAs so the
// scheduler can overlap the WMMA chain with trailing loads (partial
// s_wait_loadcnt instead of a full drain per WMMA).
// ---------------------------------------------------------------------------
template <typename TA, typename TB>
__device__ __forceinline__ void mma_32x64(const TA* __restrict__ Ab, int lda,
                                          const TB* __restrict__ Wb, int ldb,
                                          int K, int m0, int n0, int lane,
                                          v8f (&acc)[2][4]) {
  for (int k0 = 0; k0 < K; k0 += 32) {
    v16h a0 = load_a(Ab, lda, m0,      k0, lane);
    v16h a1 = load_a(Ab, lda, m0 + 16, k0, lane);
    v16h b[4];
#pragma unroll
    for (int t = 0; t < 4; ++t) b[t] = load_b(Wb, ldb, n0 + 16 * t, k0, lane);
#pragma unroll
    for (int t = 0; t < 4; ++t) {
      acc[0][t] = WMMA_F16(a0, b[t], acc[0][t]);
      acc[1][t] = WMMA_F16(a1, b[t], acc[1][t]);
    }
  }
}

// ---------------------------------------------------------------------------
// Projection GEMM:  out[m,n] = sum_k A[m,k]*W[n,k] + bias[n]
// A f32 [M x DIM], W f32 [DIM x DIM].  One wave computes a 32x64 tile.
// EPI 0: f32 token-major store.  EPI 1: f16 store as per-head V^T:
//        dst[b*DIM*LKV + n*LKV + l]  (m = b*LKV + l)
// ---------------------------------------------------------------------------
template <int EPI>
__global__ __launch_bounds__(256) void gemm_xw(
    const float* __restrict__ A, const float* __restrict__ W,
    const float* __restrict__ bias, float* __restrict__ outF,
    f16* __restrict__ outH, int M) {
  const int lane = threadIdx.x & 31;
  int gwid = (int)((blockIdx.x * blockDim.x + threadIdx.x) >> 5);
  const int tilesN = DIM / 64;
  if (gwid >= (M / 32) * tilesN) return;
  const int tn = gwid % tilesN, tm = gwid / tilesN;
  const int m0 = tm * 32, n0 = tn * 64;

  v8f acc[2][4] = {};
  mma_32x64(A, DIM, W, DIM, DIM, m0, n0, lane, acc);

  const int col = lane & 15, hl = lane >> 4;
#pragma unroll
  for (int mi = 0; mi < 2; ++mi)
#pragma unroll
    for (int t = 0; t < 4; ++t) {
      const int n = n0 + 16 * t + col;
      const float bv = bias[n];
#pragma unroll
      for (int r = 0; r < 8; ++r) {
        const int m = m0 + 16 * mi + r + 8 * hl;
        const float v = acc[mi][t][r] + bv;
        if (EPI == 0) {
          outF[(size_t)m * DIM + n] = v;
        } else {
          const int bi = m / LKV, l = m % LKV;
          outH[(size_t)bi * DIM * LKV + (size_t)n * LKV + l] = (f16)v;
        }
      }
    }
}

// ---------------------------------------------------------------------------
// Per-head LayerNorm over Hd=128, one wave per (token, head) row.
// src: f32 token-major [BB*L, DIM]; dst: f16 head-major [BB,NH,L,HD].
// ---------------------------------------------------------------------------
__global__ __launch_bounds__(256) void ln_head(
    const float* __restrict__ src, f16* __restrict__ dst, int L,
    const float* __restrict__ g, const float* __restrict__ bt) {
  const int lane = threadIdx.x & 31;
  int gwid = (int)((blockIdx.x * blockDim.x + threadIdx.x) >> 5);
  if (gwid >= BB * L * NH) return;
  const int h = gwid % NH, m = gwid / NH;
  const float* p = src + (size_t)m * DIM + h * HD;

  float x[4];
#pragma unroll
  for (int i = 0; i < 4; ++i) x[i] = p[lane + 32 * i];
  float s = x[0] + x[1] + x[2] + x[3];
#pragma unroll
  for (int o = 16; o >= 1; o >>= 1) s += __shfl_xor(s, o, 32);
  const float mean = s * (1.f / HD);
  float q = 0.f;
#pragma unroll
  for (int i = 0; i < 4; ++i) { float d = x[i] - mean; q += d * d; }
#pragma unroll
  for (int o = 16; o >= 1; o >>= 1) q += __shfl_xor(q, o, 32);
  const float rstd = rsqrtf(q * (1.f / HD) + LN_EPS);

  const int bi = m / L, pos = m % L;
  f16* d = dst + (((size_t)bi * NH + h) * L + pos) * HD;
#pragma unroll
  for (int i = 0; i < 4; ++i) {
    const int idx = lane + 32 * i;
    d[idx] = (f16)((x[i] - mean) * rstd * g[idx] + bt[idx]);
  }
}

// ---------------------------------------------------------------------------
// Scores: S[q,l] = scale * sum_d Q[q,d]*K[l,d], per (b,h) batch (blockIdx.y).
// 32x64 tile per wave, K = HD = 128 (4 k-steps).
// ---------------------------------------------------------------------------
__global__ __launch_bounds__(256) void gemm_scores(
    const f16* __restrict__ Q, const f16* __restrict__ Kc,
    float* __restrict__ S) {
  const int bh = blockIdx.y;
  const f16* Ab = Q + (size_t)bh * LQ * HD;
  const f16* Wb = Kc + (size_t)bh * LKV * HD;
  float* Ob = S + (size_t)bh * LQ * LKV;

  const int lane = threadIdx.x & 31;
  const int gwid = (int)((blockIdx.x * blockDim.x + threadIdx.x) >> 5);
  const int tilesN = LKV / 64;
  const int tn = gwid % tilesN, tm = gwid / tilesN;
  const int m0 = tm * 32, n0 = tn * 64;

  v8f acc[2][4] = {};
  mma_32x64(Ab, HD, Wb, HD, HD, m0, n0, lane, acc);

  const int col = lane & 15, hl = lane >> 4;
#pragma unroll
  for (int mi = 0; mi < 2; ++mi)
#pragma unroll
    for (int t = 0; t < 4; ++t)
#pragma unroll
      for (int r = 0; r < 8; ++r)
        Ob[(size_t)(m0 + 16 * mi + r + 8 * hl) * LKV + (n0 + 16 * t + col)] =
            acc[mi][t][r] * QK_SCALE;
}

// ---------------------------------------------------------------------------
// Row softmax in place: one 256-thread block per row of LKV=4096.
// ---------------------------------------------------------------------------
__global__ __launch_bounds__(256) void softmax_rows(float* __restrict__ S) {
  constexpr int T = 256, PER = LKV / T;
  float* p = S + (size_t)blockIdx.x * LKV;
  const int tid = threadIdx.x, lane = tid & 31, wid = tid >> 5;
  __shared__ float red[8];

  float x[PER];
  float mx = -3.4e38f;
#pragma unroll
  for (int i = 0; i < PER; ++i) { x[i] = p[tid + T * i]; mx = fmaxf(mx, x[i]); }
#pragma unroll
  for (int o = 16; o >= 1; o >>= 1) mx = fmaxf(mx, __shfl_xor(mx, o, 32));
  if (lane == 0) red[wid] = mx;
  __syncthreads();
  mx = red[0];
#pragma unroll
  for (int i = 1; i < 8; ++i) mx = fmaxf(mx, red[i]);

  float s = 0.f;
#pragma unroll
  for (int i = 0; i < PER; ++i) { x[i] = __expf(x[i] - mx); s += x[i]; }
#pragma unroll
  for (int o = 16; o >= 1; o >>= 1) s += __shfl_xor(s, o, 32);
  __syncthreads();
  if (lane == 0) red[wid] = s;
  __syncthreads();
  s = red[0];
#pragma unroll
  for (int i = 1; i < 8; ++i) s += red[i];
  const float inv = 1.f / s;
#pragma unroll
  for (int i = 0; i < PER; ++i) p[tid + T * i] = x[i] * inv;
}

// ---------------------------------------------------------------------------
// PV: O[q,hd] = sum_l P[q,l]*V[l,hd], per (b,h) batch.  P f32 (cvt on load),
// V supplied pre-transposed per head (Vt[(bh*HD+hd)*LKV + l]).  Output f32
// token-major [BB,LQ,DIM] for the O projection.
// ---------------------------------------------------------------------------
__global__ __launch_bounds__(256) void gemm_av(
    const float* __restrict__ P, const f16* __restrict__ Vt,
    float* __restrict__ O) {
  const int bh = blockIdx.y;
  const float* Ab = P + (size_t)bh * LQ * LKV;
  const f16* Wb = Vt + (size_t)bh * HD * LKV;

  const int lane = threadIdx.x & 31;
  const int gwid = (int)((blockIdx.x * blockDim.x + threadIdx.x) >> 5);
  const int tilesN = HD / 64;
  const int tn = gwid % tilesN, tm = gwid / tilesN;
  const int m0 = tm * 32, n0 = tn * 64;

  v8f acc[2][4] = {};
  mma_32x64(Ab, LKV, Wb, LKV, LKV, m0, n0, lane, acc);

  const int col = lane & 15, hl = lane >> 4;
  const int bi = bh / NH, h = bh % NH;
#pragma unroll
  for (int mi = 0; mi < 2; ++mi)
#pragma unroll
    for (int t = 0; t < 4; ++t)
#pragma unroll
      for (int r = 0; r < 8; ++r) {
        const int q = m0 + 16 * mi + r + 8 * hl;
        const int hd = n0 + 16 * t + col;
        O[((size_t)bi * LQ + q) * DIM + h * HD + hd] = acc[mi][t][r];
      }
}

// ---------------------------------------------------------------------------
extern "C" void kernel_launch(void* const* d_in, const int* in_sizes, int n_in,
                              void* d_out, int out_size, void* d_ws,
                              size_t ws_size, hipStream_t stream) {
  (void)in_sizes; (void)n_in; (void)out_size; (void)ws_size;
  const float* hidden = (const float*)d_in[0];
  const float* cross  = (const float*)d_in[1];
  const float* q_w = (const float*)d_in[2];
  const float* q_b = (const float*)d_in[3];
  const float* k_w = (const float*)d_in[4];
  const float* k_b = (const float*)d_in[5];
  const float* v_w = (const float*)d_in[6];
  const float* v_b = (const float*)d_in[7];
  const float* o_w = (const float*)d_in[8];
  const float* o_b = (const float*)d_in[9];
  const float* qn_g = (const float*)d_in[10];
  const float* qn_b = (const float*)d_in[11];
  const float* kn_g = (const float*)d_in[12];
  const float* kn_b = (const float*)d_in[13];

  // Workspace layout (256B-aligned):
  //  Q16  f16 [BB,NH,LQ,HD]  16 MiB   @ 0
  //  K16  f16 [BB,NH,LKV,HD] 32 MiB   @ 16 MiB
  //  Vt   f16 [BB,NH,HD,LKV] 32 MiB   @ 48 MiB
  //  SCR  f32 scratch        64 MiB   @ 80 MiB (pre-LN Q/K, then PV output)
  char* ws = (char*)d_ws;
  f16*   Q16 = (f16*)ws;
  f16*   K16 = (f16*)(ws + (size_t)16 * 1024 * 1024);
  f16*   Vt  = (f16*)(ws + (size_t)48 * 1024 * 1024);
  float* SCR = (float*)(ws + (size_t)80 * 1024 * 1024);

  float* outO = (float*)d_out;                                   // [2,2048,2048]
  float* outW = (float*)d_out + (size_t)BB * LQ * DIM;           // [2,16,2048,4096]

  // 1) Q = hidden @ q_w^T + q_b   -> SCR (f32)
  { int M = BB * LQ; int blk = (M / 32) * (DIM / 64) / 8;
    gemm_xw<0><<<blk, 256, 0, stream>>>(hidden, q_w, q_b, SCR, nullptr, M); }
  // 2) per-head LN(Q) -> Q16 (f16 head-major)
  ln_head<<<BB * LQ * NH / 8, 256, 0, stream>>>(SCR, Q16, LQ, qn_g, qn_b);
  // 3) K = cross @ k_w^T + k_b    -> SCR (f32)
  { int M = BB * LKV; int blk = (M / 32) * (DIM / 64) / 8;
    gemm_xw<0><<<blk, 256, 0, stream>>>(cross, k_w, k_b, SCR, nullptr, M); }
  // 4) per-head LN(K) -> K16
  ln_head<<<BB * LKV * NH / 8, 256, 0, stream>>>(SCR, K16, LKV, kn_g, kn_b);
  // 5) V = cross @ v_w^T + v_b    -> Vt (f16, per-head transposed)
  { int M = BB * LKV; int blk = (M / 32) * (DIM / 64) / 8;
    gemm_xw<1><<<blk, 256, 0, stream>>>(cross, v_w, v_b, nullptr, Vt, M); }
  // 6) scaled scores -> d_out weights region (f32)
  { dim3 g((LQ / 32) * (LKV / 64) / 8, BB * NH);
    gemm_scores<<<g, 256, 0, stream>>>(Q16, K16, outW); }
  // 7) softmax in place (this IS the attn_weights output)
  softmax_rows<<<BB * NH * LQ, 256, 0, stream>>>(outW);
  // 8) PV -> SCR (f32 token-major)
  { dim3 g((LQ / 32) * (HD / 64) / 8, BB * NH);
    gemm_av<<<g, 256, 0, stream>>>(outW, Vt, SCR); }
  // 9) attn_output = SCR @ o_w^T + o_b -> d_out
  { int M = BB * LQ; int blk = (M / 32) * (DIM / 64) / 8;
    gemm_xw<0><<<blk, 256, 0, stream>>>(SCR, o_w, o_b, outO, nullptr, M); }
}
